// ClicksMerger_24567212933948
// MI455X (gfx1250) — matrix operand order repaired and checked by reading, not compiled, not run on hardware
//
#include <hip/hip_runtime.h>

// ---------------- types for CDNA5 WMMA ----------------
typedef __attribute__((ext_vector_type(16))) __bf16 v16bf;
typedef __attribute__((ext_vector_type(8)))  __bf16 v8bf;
typedef __attribute__((ext_vector_type(8)))  float  v8f;

#define CCOL 256      // number of clicks / columns
#define KCHUNK 64     // P-rows per LDS chunk in gram kernel
#define KPAD 72       // padded k-extent (bank spread), 72*2B = 144B row stride (16B multiple)
#define NCPAD 224     // padded class dim (multiple of 32)

union FragBF { v16bf v; v8bf h[2]; };

// ---------------- zero init ----------------
__global__ void zero_kernel(float* __restrict__ p, int n) {
    int i = blockIdx.x * 256 + threadIdx.x;
    if (i < n) p[i] = 0.0f;
}

// ---------------- Gram = M^T M (bf16 WMMA, f32 acc) + column sums ----------------
__global__ __launch_bounds__(512)
void gram_kernel(const float* __restrict__ M, int P,
                 float* __restrict__ gram, float* __restrict__ s) {
    __shared__ __align__(16) __bf16 lds[CCOL * KPAD];   // 36,864 B

    const int t    = threadIdx.x;
    const int lane = t & 31;
    const int w    = t >> 5;           // wave id 0..15
    const int l16  = lane & 15;
    const int half = lane >> 4;        // 0: lanes 0-15, 1: lanes 16-31
    const int ci0  = w * 16;           // this wave's A-tile row block

    v8f acc[16];
    const v8f vzero = {0.f,0.f,0.f,0.f,0.f,0.f,0.f,0.f};
    #pragma unroll
    for (int j = 0; j < 16; ++j) acc[j] = vzero;

    const int cload = t & 255;         // column this thread loads
    const int krow  = t >> 8;          // 0 or 1
    float colsum = 0.0f;

    const int nchunks = (P + KCHUNK - 1) / KCHUNK;
    for (int chunk = blockIdx.x; chunk < nchunks; chunk += gridDim.x) {
        const long long p0 = (long long)chunk * KCHUNK;

        // stage 64 x 256 f32 -> bf16 transposed into LDS; fuse column sums
        #pragma unroll
        for (int it = 0; it < KCHUNK / 2; ++it) {
            const int k = 2 * it + krow;
            const long long p = p0 + k;
            float v = 0.0f;
            if (p < (long long)P) v = M[p * CCOL + cload];
            colsum += v;
            lds[cload * KPAD + k] = (__bf16)v;
        }
        // prefetch next chunk for this block
        {
            const long long pn = p0 + (long long)gridDim.x * KCHUNK;
            if (pn < (long long)P)
                __builtin_prefetch(&M[pn * CCOL + cload], 0, 1);
        }
        __syncthreads();

        #pragma unroll
        for (int kk = 0; kk < 2; ++kk) {
            const int k0 = kk * 32;
            // A fragment: A[m][k] = Mbf[p0+k0+k][ci0+m]
            FragBF A;
            const __bf16* abase = &lds[(ci0 + l16) * KPAD + k0 + half * 8];
            A.h[0] = *(const v8bf*)(abase);        // lanes<16: K0-7  ; lanes>=16: K8-15
            A.h[1] = *(const v8bf*)(abase + 16);   // lanes<16: K16-23; lanes>=16: K24-31
            #pragma unroll
            for (int tj = 0; tj < 16; ++tj) {
                // B fragment: B[k][n] = Mbf[p0+k0+k][tj*16+n]
                FragBF B;
                const __bf16* bbase = &lds[(tj * 16 + l16) * KPAD + k0 + half * 8];
                B.h[0] = *(const v8bf*)(bbase);       // K 0-7  (or 8-15)
                B.h[1] = *(const v8bf*)(bbase + 16);  // K 16-23 (or 24-31)
                acc[tj] = __builtin_amdgcn_wmma_f32_16x16x32_bf16(
                    false, A.v, false, B.v, (short)0, acc[tj], false, false);
            }
        }
        __syncthreads();
    }

    // accumulate into global gram
    #pragma unroll
    for (int tj = 0; tj < 16; ++tj) {
        #pragma unroll
        for (int r = 0; r < 8; ++r) {
            const int row = ci0 + half * 8 + r;
            const int col = tj * 16 + l16;
            atomicAdd(&gram[row * CCOL + col], acc[tj][r]);
        }
    }
    atomicAdd(&s[cload], colsum);
}

// ---------------- layernorm -> bf16 padded rows in workspace ----------------
__global__ void ln_kernel(const float* __restrict__ logits, int nc,
                          __bf16* __restrict__ pbf) {
    const int c = threadIdx.x;   // one row per thread, 256 threads
    float mu = 0.0f;
    for (int j = 0; j < nc; ++j) mu += logits[c * nc + j];
    mu /= (float)nc;
    float var = 0.0f;
    for (int j = 0; j < nc; ++j) { float d = logits[c * nc + j] - mu; var += d * d; }
    var /= (float)nc;
    const float rs = rsqrtf(var + 1e-5f);
    for (int j = 0; j < nc; ++j)
        pbf[c * NCPAD + j] = (__bf16)((logits[c * nc + j] - mu) * rs);
    for (int j = nc; j < NCPAD; ++j)
        pbf[c * NCPAD + j] = (__bf16)0.0f;
}

// ---------------- sim = p p^T via bf16 WMMA (single block, 16 waves) ----------------
__global__ __launch_bounds__(512)
void simw_kernel(const __bf16* __restrict__ pbf, float* __restrict__ sim) {
    const int t    = threadIdx.x;
    const int lane = t & 31;
    const int w    = t >> 5;
    const int l16  = lane & 15;
    const int half = lane >> 4;
    const int ci0  = w * 16;

    v8f acc[16];
    const v8f vzero = {0.f,0.f,0.f,0.f,0.f,0.f,0.f,0.f};
    #pragma unroll
    for (int j = 0; j < 16; ++j) acc[j] = vzero;

    for (int k0 = 0; k0 < NCPAD; k0 += 32) {
        FragBF A;
        const __bf16* abase = &pbf[(ci0 + l16) * NCPAD + k0 + half * 8];
        A.h[0] = *(const v8bf*)(abase);
        A.h[1] = *(const v8bf*)(abase + 16);
        #pragma unroll
        for (int tj = 0; tj < 16; ++tj) {
            FragBF B;
            const __bf16* bbase = &pbf[(tj * 16 + l16) * NCPAD + k0 + half * 8];
            B.h[0] = *(const v8bf*)(bbase);
            B.h[1] = *(const v8bf*)(bbase + 16);
            acc[tj] = __builtin_amdgcn_wmma_f32_16x16x32_bf16(
                false, A.v, false, B.v, (short)0, acc[tj], false, false);
        }
    }
    #pragma unroll
    for (int tj = 0; tj < 16; ++tj)
        #pragma unroll
        for (int r = 0; r < 8; ++r)
            sim[(ci0 + half * 8 + r) * CCOL + tj * 16 + l16] = acc[tj][r];
}

// ---------------- greedy association scan (1 block, 256 threads) ----------------
__global__ __launch_bounds__(256)
void assoc_kernel(const float* __restrict__ gram, const float* __restrict__ s,
                  const float* __restrict__ sim, int* __restrict__ labels,
                  float* __restrict__ out_labels, float* __restrict__ out_valid) {
    __shared__ float ss[CCOL];
    __shared__ int   lab[CCOL];
    __shared__ int   asg[CCOL];
    const int t = threadIdx.x;
    ss[t] = s[t]; lab[t] = -1; asg[t] = 0;
    __syncthreads();
    for (int i = 0; i < CCOL; ++i) {
        const float dice = 2.0f * gram[i * CCOL + t] / (ss[i] + ss[t]);
        const bool cond  = (dice > 0.4f) && (sim[i * CCOL + t] > 0.5f);
        const int ai = asg[i];
        const int at = asg[t];
        __syncthreads();
        if (cond && !ai && !at) { asg[t] = 1; lab[t] = i; }
        __syncthreads();
    }
    labels[t] = lab[t];
    int cnt = 0;
    for (int c2 = 0; c2 < CCOL; ++c2) cnt += (lab[c2] == t) ? 1 : 0;
    out_labels[t] = (float)lab[t];
    out_valid[t]  = (cnt > 0) ? 1.0f : 0.0f;
}

// ---------------- masks merge: per-row LDS scatter-max ----------------
#define MROWS 8
__global__ __launch_bounds__(256)
void mask_merge_kernel(const float* __restrict__ M, const int* __restrict__ labels,
                       float* __restrict__ outm, int P) {
    __shared__ int buf[MROWS][CCOL];
    const int t = threadIdx.x;
    const int g = labels[t];
    const long long nrb = (P + MROWS - 1) / MROWS;
    for (long long rb = blockIdx.x; rb < nrb; rb += gridDim.x) {
        const long long p0 = rb * MROWS;
        #pragma unroll
        for (int r = 0; r < MROWS; ++r) buf[r][t] = 0;  // masks >= 0: int order == float order
        __syncthreads();
        #pragma unroll
        for (int r = 0; r < MROWS; ++r) {
            const long long p = p0 + r;
            if (p < (long long)P && g >= 0)
                atomicMax(&buf[r][g], __float_as_int(M[p * CCOL + t]));
        }
        __syncthreads();
        #pragma unroll
        for (int r = 0; r < MROWS; ++r) {
            const long long p = p0 + r;
            if (p < (long long)P)
                outm[p * CCOL + t] = __int_as_float(buf[r][t]);
        }
        __syncthreads();
    }
}

// ---------------- cls merge: order-preserving uint max, sentinel 0 ----------------
__global__ __launch_bounds__(256)
void cls_merge_kernel(const float* __restrict__ logits, int nc,
                      const int* __restrict__ labels, float* __restrict__ outc) {
    __shared__ unsigned buf[CCOL];
    const int t = threadIdx.x;   // click / group index
    const int j = blockIdx.x;    // class index
    buf[t] = 0u;
    const int g = labels[t];
    const unsigned bits = __float_as_uint(logits[t * nc + j]);
    const unsigned mapped = (bits & 0x80000000u) ? ~bits : (bits | 0x80000000u);
    __syncthreads();
    if (g >= 0) atomicMax(&buf[g], mapped);
    __syncthreads();
    const unsigned u = buf[t];
    float outv = 0.0f;
    if (u != 0u) {
        const unsigned b = (u & 0x80000000u) ? (u & 0x7FFFFFFFu) : ~u;
        outv = __uint_as_float(b);
    }
    outc[t * nc + j] = outv;
}

// ---------------- host launcher ----------------
extern "C" void kernel_launch(void* const* d_in, const int* in_sizes, int n_in,
                              void* d_out, int out_size, void* d_ws, size_t ws_size,
                              hipStream_t stream) {
    const float* M      = (const float*)d_in[0];   // [P, 256]
    const float* logits = (const float*)d_in[1];   // [256, NC]
    const int P  = in_sizes[0] / CCOL;             // 200000
    const int nc = in_sizes[1] / CCOL;             // 200

    char* ws = (char*)d_ws;
    float*  gram   = (float*)(ws + 0);            // 256KB
    float*  s      = (float*)(ws + 262144);       // 1KB (contiguous after gram)
    float*  sim    = (float*)(ws + 263168);       // 256KB
    int*    labels = (int*)  (ws + 525312);       // 1KB
    __bf16* pbf    = (__bf16*)(ws + 526336);      // 112KB (16B aligned)

    float* out        = (float*)d_out;
    float* out_labels = out;                       // [256]
    float* out_valid  = out + 256;                 // [256]
    float* out_masks  = out + 512;                 // [P,256]
    float* out_cls    = out + 512 + (long long)P * CCOL;  // [256,NC]

    zero_kernel<<<(65792 + 255) / 256, 256, 0, stream>>>(gram, 65792); // gram + s
    gram_kernel<<<128, 512, 0, stream>>>(M, P, gram, s);
    ln_kernel<<<1, 256, 0, stream>>>(logits, nc, pbf);
    simw_kernel<<<1, 512, 0, stream>>>(pbf, sim);
    assoc_kernel<<<1, 256, 0, stream>>>(gram, s, sim, labels, out_labels, out_valid);
    mask_merge_kernel<<<2048, 256, 0, stream>>>(M, labels, out_masks, P);
    cls_merge_kernel<<<nc, 256, 0, stream>>>(logits, nc, labels, out_cls);
}
